// qSAC_76759655514823
// MI455X (gfx1250) — compile-verified, matrix operand order checked by reading, not run on hardware
//
#include <hip/hip_runtime.h>

// ---- types ----------------------------------------------------------------
typedef __attribute__((ext_vector_type(16))) __bf16       v16bf;
typedef __attribute__((ext_vector_type(8)))  float        v8f;
typedef __attribute__((ext_vector_type(4)))  float        f32x4;
typedef __attribute__((ext_vector_type(4)))  unsigned int u32x4;
typedef __attribute__((ext_vector_type(2)))  unsigned int u32x2;

// ---- problem / tiling constants ------------------------------------------
constexpr int IN   = 1024;   // K
constexpr int OUTN = 1024;   // N
constexpr int BM   = 128;    // block tile M
constexpr int BN   = 128;    // block tile N
constexpr int BK   = 64;     // block tile K
constexpr int LDK  = 72;     // padded LDS row stride in bf16 elems (144B, 16B-aligned)
constexpr int TILE = 128 * LDK;          // elems per LDS tile buffer
constexpr float M_SCALE = 0.00390625f;   // 2^-8

// f32 pair -> packed 2x bf16 (a low, b high), round-half-up.
// 3 VALU: two v_add_u32 (+0x8000 rounding) and one v_perm_b32 that gathers
// the two rounded high halves: result bytes = {ub[3], ub[2], ua[3], ua[2]}.
__device__ __forceinline__ unsigned pack_bf16(float a, float b) {
  const unsigned ua = __float_as_uint(a) + 0x8000u;
  const unsigned ub = __float_as_uint(b) + 0x8000u;
  return __builtin_amdgcn_perm(ub, ua, 0x07060302u);
}

// Load one 16x32 bf16 A/B fragment from an LDS tile (row-major, stride LDK).
// Lane layout per CDNA5 ISA 7.12.2: lanes 0-15 -> row r, K = kb+[0..7] and
// kb+16+[0..7]; lanes 16-31 -> row r, K = kb+8+[0..7] and kb+24+[0..7].
__device__ __forceinline__ v16bf load_frag(const unsigned short* tile,
                                           int row0, int kb, int lane) {
  const int g = (lane >> 4) & 1;
  const int r = lane & 15;
  const unsigned short* p = tile + (row0 + r) * LDK + kb + g * 8;
  union { v16bf v; u32x4 q[2]; } u;
  u.q[0] = *(const u32x4*)(p);
  u.q[1] = *(const u32x4*)(p + 16);
  return u.v;
}

__global__ __launch_bounds__(256)
void qlinear_wmma_bf16(const float* __restrict__ x, const float* __restrict__ W,
                       const float* __restrict__ bias, float* __restrict__ out) {
  // ping-pong buffers: [A0 | B0 | A1 | B1], 72KB total
  __shared__ unsigned short smem[4 * TILE];

  const int tid    = threadIdx.x;
  const int lane   = tid & 31;
  const int wave   = tid >> 5;
  const int waveM  = wave & 3;              // 4 waves along M (32 rows each)
  const int waveN  = wave >> 2;             // 2 waves along N (64 cols each)
  const int blockM = blockIdx.y * BM;
  const int blockN = blockIdx.x * BN;

  // global staging: thread covers rows lrow+16*i (i=0..7), 4 floats at lcol
  const int lrow = tid >> 4;                // 0..15
  const int lcol = (tid & 15) * 4;          // 0..60
  const float* xrow = x + (size_t)(blockM + lrow) * IN + lcol;
  const float* wrow = W + (size_t)(blockN + lrow) * IN + lcol;

  u32x2 xs[8], ws[8];                       // staged bf16-packed tiles (32 VGPRs)

  auto gload = [&](int k0) {
#pragma unroll
    for (int i = 0; i < 8; ++i) {
      f32x4 vx = *(const f32x4*)(xrow + (size_t)i * 16 * IN + k0);
      f32x4 vw = *(const f32x4*)(wrow + (size_t)i * 16 * IN + k0);
      u32x2 px; px.x = pack_bf16(vx.x, vx.y); px.y = pack_bf16(vx.z, vx.w);
      u32x2 pw; pw.x = pack_bf16(vw.x, vw.y); pw.y = pack_bf16(vw.z, vw.w);
      xs[i] = px;
      ws[i] = pw;
    }
  };

  auto sstore = [&](int buf) {
    unsigned short* A  = smem + buf * 2 * TILE;
    unsigned short* Bt = A + TILE;
#pragma unroll
    for (int i = 0; i < 8; ++i) {
      const int off = (lrow + 16 * i) * LDK + lcol;
      *(u32x2*)(A + off)  = xs[i];
      *(u32x2*)(Bt + off) = ws[i];
    }
  };

  v8f acc[2][4] = {};

  gload(0);
  sstore(0);
  __syncthreads();

  const int KT = IN / BK;                   // 16 K-steps
  for (int t = 0; t < KT; ++t) {
    const int cur = t & 1;

    if (t + 1 < KT) {
      gload((t + 1) * BK);                  // in flight during WMMA block
      if (t + 2 < KT) {                     // near-cache RT prefetch of t+2 tile
        __builtin_prefetch(xrow + (size_t)(t + 2) * BK, 0, 3);
        __builtin_prefetch(wrow + (size_t)(t + 2) * BK, 0, 3);
      }
    }

    const unsigned short* A  = smem + cur * 2 * TILE;
    const unsigned short* Bt = A + TILE;

#pragma unroll
    for (int kc = 0; kc < 2; ++kc) {
      const int kb = kc * 32;
      v16bf a0 = load_frag(A, waveM * 32,      kb, lane);
      v16bf a1 = load_frag(A, waveM * 32 + 16, kb, lane);
#pragma unroll
      for (int fn = 0; fn < 4; ++fn) {
        v16bf bb = load_frag(Bt, waveN * 64 + fn * 16, kb, lane);
        acc[0][fn] = __builtin_amdgcn_wmma_f32_16x16x32_bf16(
            false, a0, false, bb, (short)0, acc[0][fn], false, false);
        acc[1][fn] = __builtin_amdgcn_wmma_f32_16x16x32_bf16(
            false, a1, false, bb, (short)0, acc[1][fn], false, false);
      }
    }

    // buf (1-cur) was last read in iteration t-1 (barrier-protected): safe to fill
    if (t + 1 < KT) sstore(1 - cur);
    __syncthreads();
  }

  // epilogue: relu(M * (acc + bias)); C/D layout: lane = N col, VGPR j = M row
  const int r = lane & 15;
  const int g = lane >> 4;
#pragma unroll
  for (int fn = 0; fn < 4; ++fn) {
    const int n  = blockN + waveN * 64 + fn * 16 + r;
    const float bv = M_SCALE * bias[n];     // fold bias into one FMA per elem
#pragma unroll
    for (int fm = 0; fm < 2; ++fm) {
      const int m0 = blockM + waveM * 32 + fm * 16 + g * 8;
#pragma unroll
      for (int j = 0; j < 8; ++j) {
        const float v = fmaf(M_SCALE, acc[fm][fn][j], bv);
        out[(size_t)(m0 + j) * OUTN + n] = fmaxf(v, 0.0f);
      }
    }
  }
}

extern "C" void kernel_launch(void* const* d_in, const int* in_sizes, int n_in,
                              void* d_out, int out_size, void* d_ws, size_t ws_size,
                              hipStream_t stream) {
  (void)n_in; (void)out_size; (void)d_ws; (void)ws_size;
  const float* x = (const float*)d_in[0];   // [B, IN] f32
  const float* W = (const float*)d_in[1];   // [OUT, IN] f32
  const float* b = (const float*)d_in[2];   // [OUT] f32
  float* out = (float*)d_out;               // [B, OUT] f32

  const int Bdim = in_sizes[0] / IN;        // 16384
  dim3 grid(OUTN / BN, Bdim / BM);          // (8, 128)
  qlinear_wmma_bf16<<<grid, dim3(256), 0, stream>>>(x, W, b, out);
}
